// SAKECore_41326175322437
// MI455X (gfx1250) — matrix operand order repaired and checked by reading, not compiled, not run on hardware
//
#include <hip/hip_runtime.h>
#include <cmath>

typedef _Float16 h16;
typedef __attribute__((ext_vector_type(16))) _Float16 v16h;
typedef __attribute__((ext_vector_type(8)))  float    v8f;

#define DEV __device__ __forceinline__

namespace {
constexpr int NN = 25000;      // atoms
constexpr int PP = 400000;     // pairs
constexpr float EPSF = 1e-8f;
// PhysNet RBF constants: mu in [exp(-0.5), 1], K=50
constexpr double MU0d   = 0.60653065971263342360;           // exp(-0.5)
constexpr float  MU0f   = (float)MU0d;
constexpr float  MUSTEP = (float)((1.0 - MU0d) / 49.0);
constexpr double BWd    = (2.0 / 50.0) * (1.0 - MU0d);
constexpr float  BETA   = (float)(1.0 / (BWd * BWd));
}

DEV float silu_f(float x){ return x / (1.f + __expf(-x)); }
DEV float celu2_f(float x){ return fmaxf(x, 0.f) + fminf(0.f, 2.f * (__expf(0.5f * x) - 1.f)); }

DEV void atomicMaxF(float* addr, float val){
  unsigned int* ua = (unsigned int*)addr;
  unsigned int old = *ua;
  while (__uint_as_float(old) < val){
    unsigned int assumed = old;
    old = atomicCAS(ua, assumed, __float_as_uint(val));
    if (old == assumed) break;
  }
}

// ---- WMMA fragment loaders (CDNA5 ISA 7.12.2 / 7.12.4 layouts, wave32) ----
// A: 16x32 f16 tile from row-major LDS [16][ld], K window at k0.
// Per-lane data is two contiguous 8-half (16B) runs -> ds_load_b128 pairs.
DEV v16h frag_A(const h16* s, int ld, int k0, int lane){
  const int m = lane & 15, hf = lane >> 4;
  const h16* p = s + m * ld + k0 + hf * 8;
  v16h a;
  #pragma unroll
  for (int t = 0; t < 8; ++t) a[t] = p[t];         // K = hf*8 + 0..7
  #pragma unroll
  for (int t = 0; t < 8; ++t) a[8 + t] = p[16 + t]; // K = 16 + hf*8 + 0..7
  return a;
}
// B: weights pre-swizzled to fragment order; one aligned 32B contiguous read.
// storage: ((kb*nb + nbi)*32 + lane)*16 + t, with k = kb*32+(lane>>4)*8+(t&7)+(t>>3)*16,
// n = nbi*16+(lane&15)   (CDNA5 B-matrix 32x16 layout)
DEV v16h frag_B_swz(const h16* s, int kbi, int nbi, int nb, int lane){
  return *(const v16h*)(s + (((kbi * nb + nbi) * 32 + lane) << 4));
}
DEV v8f wmma(v16h a, v16h b, v8f c){
  return __builtin_amdgcn_wmma_f32_16x16x32_f16(false, a, false, b, (short)0, c, false, false);
}
// C/D tile: element (m = r + 8*(lane>>4), n = lane&15) in register r.

// vectorized global->LDS copy (16B chunks); halves must be a multiple of 8
DEV void cp16(h16* dst, const h16* src, int halves, int tid, int nthr){
  uint4* d = (uint4*)dst;
  const uint4* s = (const uint4*)src;
  const int n = halves >> 3;
  for (int t = tid; t < n; t += nthr) d[t] = s[t];
}

// ---------------------------------------------------------------- utilities
__global__ void k_fill(float* p, long long n, float v){
  long long t = (long long)blockIdx.x * blockDim.x + threadIdx.x;
  long long s = (long long)gridDim.x * blockDim.x;
  for (; t < n; t += s) p[t] = v;
}

// f32 weight [srows][scols] -> f16 swizzled B-fragment layout, zero padded.
__global__ void k_cvt_swz(const float* __restrict__ src, h16* __restrict__ dst,
                          int kb, int nb, int srows, int scols){
  int u = blockIdx.x * blockDim.x + threadIdx.x;
  if (u >= kb * nb * 512) return;
  int t = u & 15, lane = (u >> 4) & 31, blk = u >> 9;
  int nbi = blk % nb, kbi = blk / nb;
  int n = nbi * 16 + (lane & 15);
  int k = kbi * 32 + (lane >> 4) * 8 + (t & 7) + (t >> 3) * 16;
  dst[u] = (k < srows && n < scols) ? (h16)src[k * scols + n] : (h16)0.f;
}

// ------------------------------------------------------------- edge kernel
// Fused: geometry, h_cat gather, ein GEMM, RBF filter, eo1 GEMM+silu,
// eo2 GEMM, att GEMM+celu, segment-max(att). 2 waves/block, 16 pairs/wave.
__global__ void __launch_bounds__(64)
k_edge(const float* __restrict__ h, const float* __restrict__ x,
       const int* __restrict__ idx_i, const int* __restrict__ idx_j,
       const float* __restrict__ b_ein, const float* __restrict__ b_eo1,
       const float* __restrict__ b_eo2, const float* __restrict__ b_att,
       const h16* __restrict__ gWein,   // swz kb=4 nb=4
       const h16* __restrict__ gWeo1,   // swz kb=6 nb=4
       const h16* __restrict__ gWeo2,   // swz kb=2 nb=2
       const h16* __restrict__ gWatt,   // swz kb=1 nb=1
       float* __restrict__ edge_out,    // [P][32]
       float* __restrict__ att_out,     // [P][4]
       float* __restrict__ att_max)     // [N][4]
{
  __shared__ __align__(32) h16 sWein[128 * 64];
  __shared__ __align__(32) h16 sWeo1[192 * 64];
  __shared__ __align__(32) h16 sWeo2[64 * 32];
  __shared__ __align__(32) h16 sWatt[32 * 16];
  __shared__ __align__(16) h16 sA[2][16][192];
  __shared__ int sI[2][16], sJ[2][16];
  __shared__ float sD[2][16];

  const int tid = threadIdx.x, lane = tid & 31, w = tid >> 5;
  cp16(sWein, gWein, 128 * 64, tid, 64);
  cp16(sWeo1, gWeo1, 192 * 64, tid, 64);
  cp16(sWeo2, gWeo2, 64 * 32, tid, 64);
  cp16(sWatt, gWatt, 32 * 16, tid, 64);

  const int p0 = (blockIdx.x * 2 + w) * 16;
  // prefetch next tile's pair indices into cache
  if (lane == 0 && p0 + 32 < PP){
    __builtin_prefetch(idx_i + p0 + 32, 0, 1);
    __builtin_prefetch(idx_j + p0 + 32, 0, 1);
  }
  if (lane < 16){
    int p = p0 + lane; if (p >= PP) p = PP - 1;
    int i = idx_i[p], j = idx_j[p];
    float rx = x[j*3+0]-x[i*3+0], ry = x[j*3+1]-x[i*3+1], rz = x[j*3+2]-x[i*3+2];
    sI[w][lane] = i; sJ[w][lane] = j;
    sD[w][lane] = sqrtf(rx*rx + ry*ry + rz*rz + EPSF);
  }
  __syncthreads();

  // stage h_cat = [h_i | h_j] as f16 rows of A
  for (int q = lane; q < 16 * 128; q += 32){
    int m = q >> 7, c = q & 127;
    int a = (c < 64) ? sI[w][m] : sJ[w][m];
    sA[w][m][c] = (h16)h[a * 64 + (c & 63)];
  }

  // GEMM1: h_cat @ W_ein  (K=128, N=64 padded from 50)
  v8f acc1[4] = {};
  #pragma unroll
  for (int ks = 0; ks < 4; ++ks){
    v16h a = frag_A(&sA[w][0][0], 192, ks * 32, lane);
    #pragma unroll
    for (int nt = 0; nt < 4; ++nt)
      acc1[nt] = wmma(a, frag_B_swz(sWein, ks, nt, 4, lane), acc1[nt]);
  }
  // filtered = rbf(d) * (. + b_ein); append d; write cols 128..191 of A
  {
    const int n = lane & 15, hf = lane >> 4;
    #pragma unroll
    for (int nt = 0; nt < 4; ++nt)
      #pragma unroll
      for (int r = 0; r < 8; ++r){
        int k = nt * 16 + n, m = r + 8 * hf;
        float dm = sD[w][m], out;
        if (k < 50){
          float mu = MU0f + (float)k * MUSTEP;
          float t  = __expf(-dm) - mu;
          out = __expf(-BETA * t * t) * (acc1[nt][r] + b_ein[k]);
        } else if (k == 50) out = dm;   // d / SCALE, SCALE = 1
        else out = 0.f;
        sA[w][m][128 + k] = (h16)out;
      }
  }

  // GEMM2: edge_in @ W_eo1 (K=192 padded from 179, N=64) -> silu -> A cols 0..63
  v8f acc2[4] = {};
  #pragma unroll
  for (int ks = 0; ks < 6; ++ks){
    v16h a = frag_A(&sA[w][0][0], 192, ks * 32, lane);
    #pragma unroll
    for (int nt = 0; nt < 4; ++nt)
      acc2[nt] = wmma(a, frag_B_swz(sWeo1, ks, nt, 4, lane), acc2[nt]);
  }
  {
    const int n = lane & 15, hf = lane >> 4;
    #pragma unroll
    for (int nt = 0; nt < 4; ++nt)
      #pragma unroll
      for (int r = 0; r < 8; ++r){
        int nn = nt * 16 + n, m = r + 8 * hf;
        sA[w][m][nn] = (h16)silu_f(acc2[nt][r] + b_eo1[nn]);
      }
  }

  // GEMM3: @ W_eo2 (K=64, N=32) -> h_ij_edge; store + restage f16 in cols 0..31
  v8f acc3[2] = {};
  #pragma unroll
  for (int ks = 0; ks < 2; ++ks){
    v16h a = frag_A(&sA[w][0][0], 192, ks * 32, lane);
    #pragma unroll
    for (int nt = 0; nt < 2; ++nt)
      acc3[nt] = wmma(a, frag_B_swz(sWeo2, ks, nt, 2, lane), acc3[nt]);
  }
  {
    const int n = lane & 15, hf = lane >> 4;
    #pragma unroll
    for (int nt = 0; nt < 2; ++nt)
      #pragma unroll
      for (int r = 0; r < 8; ++r){
        int e = nt * 16 + n, m = r + 8 * hf, p = p0 + m;
        float val = acc3[nt][r] + b_eo2[e];
        if (p < PP) edge_out[p * 32 + e] = val;
        sA[w][m][e] = (h16)val;
      }
  }

  // GEMM4: @ W_att (K=32, N=16 padded from 4) -> celu(alpha=2); segment max
  v8f acc4 = {};
  {
    v16h a = frag_A(&sA[w][0][0], 192, 0, lane);
    acc4 = wmma(a, frag_B_swz(sWatt, 0, 0, 1, lane), acc4);
  }
  {
    const int n = lane & 15, hf = lane >> 4;
    if (n < 4)
      #pragma unroll
      for (int r = 0; r < 8; ++r){
        int m = r + 8 * hf, p = p0 + m;
        if (p < PP){
          float cel = celu2_f(acc4[r] + b_att[n]);
          att_out[p * 4 + n] = cel;
          atomicMaxF(&att_max[sI[w][m] * 4 + n], cel);
        }
      }
  }
}

// per (pair,head): e = exp(att - max); denom += e
__global__ void k_exp(const int* __restrict__ idx_i, float* __restrict__ att_e,
                      float* __restrict__ denom, const float* __restrict__ att_max){
  int t = blockIdx.x * blockDim.x + threadIdx.x;
  if (t >= PP * 4) return;
  int p = t >> 2, hh = t & 3, i = idx_i[p];
  float m = att_max[i * 4 + hh];
  if (!(m > -1e37f)) m = 0.f;
  float e = __expf(att_e[t] - m);
  att_e[t] = e;
  atomicAdd(&denom[i * 4 + hh], e);
}

// Fused: att_w, h_ij_sem (+seg_sum), xm GEMM, tanh, comb (+seg accum), v_ij (+seg accum)
__global__ void __launch_bounds__(128)
k_sem(const float* __restrict__ x,
      const int* __restrict__ idx_i, const int* __restrict__ idx_j,
      const float* __restrict__ edge, const float* __restrict__ att_e,
      const float* __restrict__ denom, const float* __restrict__ W_vm,
      const h16* __restrict__ gWxm,              // swz kb=4 nb=2
      float* __restrict__ h_i_sem,               // [N][128]
      float* __restrict__ comb_sum,              // [N][32][3]
      float* __restrict__ dv_sum,                // [N][3]
      float* __restrict__ cnt)                   // [N]
{
  __shared__ __align__(32) h16 sWxm[128 * 32];
  __shared__ __align__(16) h16 sA[4][16][128];
  __shared__ int sI[4][16];
  __shared__ float sDir[4][16][3];
  __shared__ float sRed[4][16];

  const int tid = threadIdx.x, lane = tid & 31, w = tid >> 5;
  cp16(sWxm, gWxm, 128 * 32, tid, 128);

  const int p0 = (blockIdx.x * 4 + w) * 16;
  if (lane == 0 && p0 + 64 < PP){
    __builtin_prefetch(idx_i + p0 + 64, 0, 1);
    __builtin_prefetch(idx_j + p0 + 64, 0, 1);
  }
  if (lane < 16){
    int p = p0 + lane; if (p >= PP) p = PP - 1;
    int i = idx_i[p], j = idx_j[p];
    float rx = x[j*3+0]-x[i*3+0], ry = x[j*3+1]-x[i*3+1], rz = x[j*3+2]-x[i*3+2];
    float d = sqrtf(rx*rx + ry*ry + rz*rz + EPSF);
    float inv = 1.f / (d + EPSF);
    sI[w][lane] = i;
    sDir[w][lane][0] = rx * inv; sDir[w][lane][1] = ry * inv; sDir[w][lane][2] = rz * inv;
    sRed[w][lane] = 0.f;
  }
  __syncthreads();

  // h_ij_sem[m][e*4+hh] = edge[e] * att_w[hh]; accumulate seg_sum with atomics
  for (int q = lane; q < 16 * 128; q += 32){
    int m = q >> 7, c = q & 127, e = c >> 2, hh = c & 3;
    int p = p0 + m; if (p >= PP) p = PP - 1;
    int i = sI[w][m];
    float aw  = att_e[p * 4 + hh] / denom[i * 4 + hh];
    float val = edge[p * 32 + e] * aw;
    sA[w][m][c] = (h16)val;
    if (p0 + m < PP) atomicAdd(&h_i_sem[i * 128 + c], val);
  }

  // GEMM: h_ij_sem @ W_xm (K=128, N=32)
  v8f acc[2] = {};
  #pragma unroll
  for (int ks = 0; ks < 4; ++ks){
    v16h a = frag_A(&sA[w][0][0], 128, ks * 32, lane);
    #pragma unroll
    for (int nt = 0; nt < 2; ++nt)
      acc[nt] = wmma(a, frag_B_swz(sWxm, ks, nt, 2, lane), acc[nt]);
  }
  {
    const int n = lane & 15, hf = lane >> 4;
    #pragma unroll
    for (int r = 0; r < 8; ++r){
      int m = r + 8 * hf;
      if (p0 + m < PP){
        int i = sI[w][m];
        float dot = 0.f;
        #pragma unroll
        for (int nt = 0; nt < 2; ++nt){
          int c = nt * 16 + n;
          float t = tanhf(acc[nt][r]);
          #pragma unroll
          for (int xx = 0; xx < 3; ++xx)
            atomicAdd(&comb_sum[i * 96 + c * 3 + xx], t * sDir[w][m][xx]);
          dot += t * W_vm[c];
        }
        atomicAdd(&sRed[w][m], dot);   // reduce over the 32 C-columns
      }
    }
  }
  __syncthreads();
  if (lane < 16 && p0 + lane < PP){
    int i = sI[w][lane];
    float dv = sRed[w][lane];
    #pragma unroll
    for (int xx = 0; xx < 3; ++xx)
      atomicAdd(&dv_sum[i * 3 + xx], dv * sDir[w][lane][xx]);
    atomicAdd(&cnt[i], 1.f);
  }
}

// spatial MLP: comb_norm2 -> silu(pn1) -> silu(pn2)
__global__ void __launch_bounds__(128)
k_spatial(const float* __restrict__ comb_sum, const float* __restrict__ cnt,
          const float* __restrict__ b_pn1, const float* __restrict__ b_pn2,
          const h16* __restrict__ gWpn1,   // swz kb=1 nb=4
          const h16* __restrict__ gWpn2,   // swz kb=2 nb=4
          float* __restrict__ h_sp)        // [N][64]
{
  __shared__ __align__(32) h16 sW1[32 * 64];
  __shared__ __align__(32) h16 sW2[64 * 64];
  __shared__ __align__(16) h16 sA[4][16][64];
  __shared__ int sI[4][16];

  const int tid = threadIdx.x, lane = tid & 31, w = tid >> 5;
  cp16(sW1, gWpn1, 32 * 64, tid, 128);
  cp16(sW2, gWpn2, 64 * 64, tid, 128);

  const int a0 = (blockIdx.x * 4 + w) * 16;
  if (lane < 16){ int a = a0 + lane; sI[w][lane] = (a < NN) ? a : NN - 1; }
  __syncthreads();

  for (int q = lane; q < 16 * 32; q += 32){
    int m = q >> 5, c = q & 31, a = sI[w][m];
    float cn = fmaxf(cnt[a], 1.f), s = 0.f;
    #pragma unroll
    for (int xx = 0; xx < 3; ++xx){
      float t = comb_sum[a * 96 + c * 3 + xx] / cn;
      s += t * t;
    }
    sA[w][m][c] = (h16)s;
  }

  v8f acc[4] = {};
  {
    v16h a = frag_A(&sA[w][0][0], 64, 0, lane);     // K = 32
    #pragma unroll
    for (int nt = 0; nt < 4; ++nt)
      acc[nt] = wmma(a, frag_B_swz(sW1, 0, nt, 4, lane), acc[nt]);
  }
  {
    const int n = lane & 15, hf = lane >> 4;
    #pragma unroll
    for (int nt = 0; nt < 4; ++nt)
      #pragma unroll
      for (int r = 0; r < 8; ++r){
        int nn = nt * 16 + n, m = r + 8 * hf;
        sA[w][m][nn] = (h16)silu_f(acc[nt][r] + b_pn1[nn]);
      }
  }
  v8f acc2[4] = {};
  #pragma unroll
  for (int ks = 0; ks < 2; ++ks){
    v16h a = frag_A(&sA[w][0][0], 64, ks * 32, lane);
    #pragma unroll
    for (int nt = 0; nt < 4; ++nt)
      acc2[nt] = wmma(a, frag_B_swz(sW2, ks, nt, 4, lane), acc2[nt]);
  }
  {
    const int n = lane & 15, hf = lane >> 4;
    #pragma unroll
    for (int nt = 0; nt < 4; ++nt)
      #pragma unroll
      for (int r = 0; r < 8; ++r){
        int nn = nt * 16 + n, m = r + 8 * hf;
        if (a0 + m < NN) h_sp[sI[w][m] * 64 + nn] = silu_f(acc2[nt][r] + b_pn2[nn]);
      }
  }
}

// node update: h_new = h + silu(silu([h|h_i_sem|h_sp] @ Wn1 + b) @ Wn2 + b)
__global__ void __launch_bounds__(64)
k_node(const float* __restrict__ h, const float* __restrict__ h_i_sem,
       const float* __restrict__ h_sp,
       const float* __restrict__ b_n1, const float* __restrict__ b_n2,
       const h16* __restrict__ gWn1,   // swz kb=8 nb=4
       const h16* __restrict__ gWn2,   // swz kb=2 nb=4
       float* __restrict__ out_h, h16* __restrict__ h_new_h)
{
  __shared__ __align__(32) h16 sW1[256 * 64];
  __shared__ __align__(32) h16 sW2[64 * 64];
  __shared__ __align__(16) h16 sA[2][16][256];
  __shared__ int sI[2][16];

  const int tid = threadIdx.x, lane = tid & 31, w = tid >> 5;
  cp16(sW1, gWn1, 256 * 64, tid, 64);
  cp16(sW2, gWn2, 64 * 64, tid, 64);

  const int a0 = (blockIdx.x * 2 + w) * 16;
  if (lane < 16){ int a = a0 + lane; sI[w][lane] = (a < NN) ? a : NN - 1; }
  __syncthreads();

  for (int q = lane; q < 16 * 256; q += 32){
    int m = q >> 8, c = q & 255, a = sI[w][m];
    float v = (c < 64) ? h[a * 64 + c]
            : (c < 192) ? h_i_sem[a * 128 + (c - 64)]
                        : h_sp[a * 64 + (c - 192)];
    sA[w][m][c] = (h16)v;
  }

  v8f acc[4] = {};
  #pragma unroll
  for (int ks = 0; ks < 8; ++ks){
    v16h a = frag_A(&sA[w][0][0], 256, ks * 32, lane);
    #pragma unroll
    for (int nt = 0; nt < 4; ++nt)
      acc[nt] = wmma(a, frag_B_swz(sW1, ks, nt, 4, lane), acc[nt]);
  }
  {
    const int n = lane & 15, hf = lane >> 4;
    #pragma unroll
    for (int nt = 0; nt < 4; ++nt)
      #pragma unroll
      for (int r = 0; r < 8; ++r){
        int nn = nt * 16 + n, m = r + 8 * hf;
        sA[w][m][nn] = (h16)silu_f(acc[nt][r] + b_n1[nn]);
      }
  }
  v8f acc2[4] = {};
  #pragma unroll
  for (int ks = 0; ks < 2; ++ks){
    v16h a = frag_A(&sA[w][0][0], 256, ks * 32, lane);
    #pragma unroll
    for (int nt = 0; nt < 4; ++nt)
      acc2[nt] = wmma(a, frag_B_swz(sW2, ks, nt, 4, lane), acc2[nt]);
  }
  {
    const int n = lane & 15, hf = lane >> 4;
    #pragma unroll
    for (int nt = 0; nt < 4; ++nt)
      #pragma unroll
      for (int r = 0; r < 8; ++r){
        int nn = nt * 16 + n, m = r + 8 * hf;
        if (a0 + m < NN){
          int a = sI[w][m];
          float val = h[a * 64 + nn] + silu_f(acc2[nt][r] + b_n2[nn]);
          out_h[a * 64 + nn] = val;
          h_new_h[a * 64 + nn] = (h16)val;
        }
      }
  }
}

// velocity: gate = 2*sigmoid(silu(h_new@Wv1+b)@Wv2); v' = gate*v + dv; x' = x + v'
__global__ void __launch_bounds__(128)
k_vel(const h16* __restrict__ h_new_h,
      const float* __restrict__ xin, const float* __restrict__ vin,
      const float* __restrict__ b_v1, const float* __restrict__ W_v2,
      const float* __restrict__ dv_sum, const float* __restrict__ cnt,
      const h16* __restrict__ gWv1,    // swz kb=2 nb=4
      float* __restrict__ out_x, float* __restrict__ out_v)
{
  __shared__ __align__(32) h16 sW1[64 * 64];
  __shared__ __align__(16) h16 sA[4][16][64];
  __shared__ int sI[4][16];
  __shared__ float sRed[4][16];

  const int tid = threadIdx.x, lane = tid & 31, w = tid >> 5;
  cp16(sW1, gWv1, 64 * 64, tid, 128);

  const int a0 = (blockIdx.x * 4 + w) * 16;
  if (lane < 16){
    int a = a0 + lane;
    sI[w][lane] = (a < NN) ? a : NN - 1;
    sRed[w][lane] = 0.f;
  }
  __syncthreads();

  for (int q = lane; q < 16 * 64; q += 32){
    int m = q >> 6, c = q & 63;
    sA[w][m][c] = h_new_h[sI[w][m] * 64 + c];
  }

  v8f acc[4] = {};
  #pragma unroll
  for (int ks = 0; ks < 2; ++ks){
    v16h a = frag_A(&sA[w][0][0], 64, ks * 32, lane);
    #pragma unroll
    for (int nt = 0; nt < 4; ++nt)
      acc[nt] = wmma(a, frag_B_swz(sW1, ks, nt, 4, lane), acc[nt]);
  }
  {
    const int n = lane & 15, hf = lane >> 4;
    #pragma unroll
    for (int r = 0; r < 8; ++r){
      int m = r + 8 * hf;
      float part = 0.f;
      #pragma unroll
      for (int nt = 0; nt < 4; ++nt){
        int nn = nt * 16 + n;
        part += silu_f(acc[nt][r] + b_v1[nn]) * W_v2[nn];
      }
      atomicAdd(&sRed[w][m], part);
    }
  }
  __syncthreads();
  if (lane < 16 && a0 + lane < NN){
    int a = sI[w][lane];
    float gate = 2.f / (1.f + __expf(-sRed[w][lane]));
    float cn = fmaxf(cnt[a], 1.f);
    #pragma unroll
    for (int xx = 0; xx < 3; ++xx){
      float dv = dv_sum[a * 3 + xx] / cn;
      float vn = gate * vin[a * 3 + xx] + dv;
      out_v[a * 3 + xx] = vn;
      out_x[a * 3 + xx] = xin[a * 3 + xx] + vn;
    }
  }
}

// ------------------------------------------------------------------ launch
extern "C" void kernel_launch(void* const* d_in, const int* in_sizes, int n_in,
                              void* d_out, int out_size, void* d_ws, size_t ws_size,
                              hipStream_t stream)
{
  (void)in_sizes; (void)n_in; (void)out_size; (void)ws_size;
  const float* h    = (const float*)d_in[0];
  const float* x    = (const float*)d_in[1];
  const float* v    = (const float*)d_in[2];
  const int*  idx_i = (const int*)d_in[3];
  const int*  idx_j = (const int*)d_in[4];
  const float* W_ein=(const float*)d_in[5],  *b_ein=(const float*)d_in[6];
  const float* W_eo1=(const float*)d_in[7],  *b_eo1=(const float*)d_in[8];
  const float* W_eo2=(const float*)d_in[9],  *b_eo2=(const float*)d_in[10];
  const float* W_att=(const float*)d_in[11], *b_att=(const float*)d_in[12];
  const float* W_xm =(const float*)d_in[13], *W_vm =(const float*)d_in[14];
  const float* W_pn1=(const float*)d_in[15], *b_pn1=(const float*)d_in[16];
  const float* W_pn2=(const float*)d_in[17], *b_pn2=(const float*)d_in[18];
  const float* W_n1 =(const float*)d_in[19], *b_n1 =(const float*)d_in[20];
  const float* W_n2 =(const float*)d_in[21], *b_n2 =(const float*)d_in[22];
  const float* W_v1 =(const float*)d_in[23], *b_v1 =(const float*)d_in[24];
  const float* W_v2 =(const float*)d_in[25];

  char* ws = (char*)d_ws;
  size_t o = 0;
  auto alloc = [&](size_t bytes) -> char* {
    char* p = ws + o; o += (bytes + 255) & ~(size_t)255; return p;
  };
  float* edge   = (float*)alloc(sizeof(float) * (size_t)PP * 32);
  float* att_e  = (float*)alloc(sizeof(float) * (size_t)PP * 4);
  float* attmax = (float*)alloc(sizeof(float) * (size_t)NN * 4);
  size_t z0 = o;
  float* denom  = (float*)alloc(sizeof(float) * (size_t)NN * 4);
  float* hsem   = (float*)alloc(sizeof(float) * (size_t)NN * 128);
  float* combs  = (float*)alloc(sizeof(float) * (size_t)NN * 96);
  float* dvs    = (float*)alloc(sizeof(float) * (size_t)NN * 3);
  float* cnt    = (float*)alloc(sizeof(float) * (size_t)NN);
  size_t z1 = o;
  float* hsp    = (float*)alloc(sizeof(float) * (size_t)NN * 64);
  h16*  hnewh   = (h16*)alloc(sizeof(h16) * (size_t)NN * 64);
  h16*  Wein_h  = (h16*)alloc(sizeof(h16) * 128 * 64);
  h16*  Weo1_h  = (h16*)alloc(sizeof(h16) * 192 * 64);
  h16*  Weo2_h  = (h16*)alloc(sizeof(h16) * 64 * 32);
  h16*  Watt_h  = (h16*)alloc(sizeof(h16) * 32 * 16);
  h16*  Wxm_h   = (h16*)alloc(sizeof(h16) * 128 * 32);
  h16*  Wpn1_h  = (h16*)alloc(sizeof(h16) * 32 * 64);
  h16*  Wpn2_h  = (h16*)alloc(sizeof(h16) * 64 * 64);
  h16*  Wn1_h   = (h16*)alloc(sizeof(h16) * 256 * 64);
  h16*  Wn2_h   = (h16*)alloc(sizeof(h16) * 64 * 64);
  h16*  Wv1_h   = (h16*)alloc(sizeof(h16) * 64 * 64);

  float* out_h = (float*)d_out;
  float* out_x = out_h + (size_t)NN * 64;
  float* out_v = out_x + (size_t)NN * 3;

  // init accumulators
  k_fill<<<256, 256, 0, stream>>>(attmax, (long long)NN * 4, -3.0e38f);
  k_fill<<<2048, 256, 0, stream>>>((float*)(ws + z0), (long long)((z1 - z0) / 4), 0.f);

  // weight conversion: f32 -> f16 swizzled into WMMA B-fragment order
  auto cvt = [&](const float* src, h16* dst, int kb, int nb, int sr, int sc){
    int tot = kb * nb * 512;
    k_cvt_swz<<<(tot + 255) / 256, 256, 0, stream>>>(src, dst, kb, nb, sr, sc);
  };
  cvt(W_ein, Wein_h, 4, 4, 128, 50);
  cvt(W_eo1, Weo1_h, 6, 4, 179, 64);
  cvt(W_eo2, Weo2_h, 2, 2, 64, 32);
  cvt(W_att, Watt_h, 1, 1, 32, 4);
  cvt(W_xm,  Wxm_h,  4, 2, 128, 32);
  cvt(W_pn1, Wpn1_h, 1, 4, 32, 64);
  cvt(W_pn2, Wpn2_h, 2, 4, 64, 64);
  cvt(W_n1,  Wn1_h,  8, 4, 256, 64);
  cvt(W_n2,  Wn2_h,  2, 4, 64, 64);
  cvt(W_v1,  Wv1_h,  2, 4, 64, 64);

  const int TP = (PP + 15) / 16;   // 25000 pair tiles
  const int TA = (NN + 15) / 16;   // 1563 atom tiles

  k_edge<<<(TP + 1) / 2, 64, 0, stream>>>(h, x, idx_i, idx_j,
      b_ein, b_eo1, b_eo2, b_att, Wein_h, Weo1_h, Weo2_h, Watt_h,
      edge, att_e, attmax);
  k_exp<<<(PP * 4 + 255) / 256, 256, 0, stream>>>(idx_i, att_e, denom, attmax);
  k_sem<<<(TP + 3) / 4, 128, 0, stream>>>(x, idx_i, idx_j, edge, att_e, denom,
      W_vm, Wxm_h, hsem, combs, dvs, cnt);
  k_spatial<<<(TA + 3) / 4, 128, 0, stream>>>(combs, cnt, b_pn1, b_pn2,
      Wpn1_h, Wpn2_h, hsp);
  k_node<<<(TA + 1) / 2, 64, 0, stream>>>(h, hsem, hsp, b_n1, b_n2,
      Wn1_h, Wn2_h, out_h, hnewh);
  k_vel<<<(TA + 3) / 4, 128, 0, stream>>>(hnewh, x, v, b_v1, W_v2,
      dvs, cnt, Wv1_h, out_x, out_v);
}
